// SimpleGAT_28965259444795
// MI455X (gfx1250) — matrix operand order repaired and checked by reading, not compiled, not run on hardware
//
#include <hip/hip_runtime.h>

// ---------------------------------------------------------------------------
// Types for CDNA5 WMMA (wave32)
// ---------------------------------------------------------------------------
typedef __attribute__((ext_vector_type(16))) _Float16 v16h;
typedef __attribute__((ext_vector_type(8)))  _Float16 v8h;
typedef __attribute__((ext_vector_type(8)))  float    v8f;

#define LDB 136   // 128 K-values per chunk + 8 halfs pad (bank decorrelation)

// ---------------------------------------------------------------------------
// f32 -> f16 conversion
// ---------------------------------------------------------------------------
__global__ void f32_to_f16_k(const float* __restrict__ in, _Float16* __restrict__ out, int n) {
    int i = blockIdx.x * blockDim.x + threadIdx.x;
    if (i < n) out[i] = (_Float16)in[i];
}

// ---------------------------------------------------------------------------
// WMMA GEMM: C[M,128] = A[M,K] * B[K,128]   (A,B f16; C f32 accumulate)
// MODE 0: += bias, ReLU, store f16 (node init layer)
// MODE 1: raw f32 store (feat = h @ Wfc)
// One wave computes a 16-row x 128-col strip (8 accumulators).
// B is staged transposed into LDS in chunks of 128 K-values.
// ---------------------------------------------------------------------------
template <int MODE>
__global__ __launch_bounds__(256) void gemm_wmma_k(
    const _Float16* __restrict__ A, const _Float16* __restrict__ B,
    const float* __restrict__ bias, void* __restrict__ Cout, int M, int K) {
    __shared__ _Float16 bt[128 * LDB];   // bt[n*LDB + kk], 34.8 KB

    const int lane  = threadIdx.x & 31;
    const int wave  = threadIdx.x >> 5;
    const int ntile = M >> 4;
    const int tile  = blockIdx.x * 8 + wave;
    const bool valid = tile < ntile;
    const int row0 = (valid ? tile : (ntile - 1)) << 4;

    const int mrow = lane & 15;              // A: row within tile / B: column within tile
    const int ksel = (lane & 16) ? 8 : 0;    // A-fragment K select (ISA 16-bit A layout)
    const int bsel = (lane & 16) ? 8 : 0;    // B-fragment K select

    v8f acc[8];
#pragma unroll
    for (int i = 0; i < 8; ++i) acc[i] = {};

    for (int kc = 0; kc < K; kc += 128) {
        __syncthreads();
        // stage B chunk transposed: B[(kc+kk)*128 + n] -> bt[n*LDB + kk]
        for (int t = threadIdx.x; t < 128 * 128; t += 256) {
            int kk = t >> 7, n = t & 127;
            bt[n * LDB + kk] = B[(size_t)(kc + kk) * 128 + n];
        }
        __syncthreads();

#pragma unroll
        for (int kk0 = 0; kk0 < 128; kk0 += 32) {
            const _Float16* pa = A + (size_t)(row0 + mrow) * K + kc + kk0 + ksel;
            if (kk0 == 0) __builtin_prefetch((const void*)(pa + 128), 0, 0);
            v16h af;
            ((v8h*)&af)[0] = *(const v8h*)(pa);
            ((v8h*)&af)[1] = *(const v8h*)(pa + 16);
#pragma unroll
            for (int nt = 0; nt < 8; ++nt) {
                const _Float16* pb = &bt[(nt * 16 + mrow) * LDB + kk0 + bsel];
                v16h bf;
                ((v8h*)&bf)[0] = *(const v8h*)(pb);
                ((v8h*)&bf)[1] = *(const v8h*)(pb + 16);
                acc[nt] = __builtin_amdgcn_wmma_f32_16x16x32_f16(
                    false, af, false, bf, (short)0, acc[nt], false, false);
            }
        }
    }

    if (!valid) return;
    // C/D layout: lane 0-15 -> rows r+0..7, lane 16-31 -> rows r+8..15; col = nt*16 + (lane&15)
    const int rbase = row0 + ((lane & 16) ? 8 : 0);
#pragma unroll
    for (int nt = 0; nt < 8; ++nt) {
        const int col = nt * 16 + (lane & 15);
        const float bv = (MODE == 0) ? bias[col] : 0.0f;
#pragma unroll
        for (int r = 0; r < 8; ++r) {
            float v = acc[nt][r] + bv;
            if (MODE == 0) {
                v = v > 0.0f ? v : 0.0f;
                ((_Float16*)Cout)[(size_t)(rbase + r) * 128 + col] = (_Float16)v;
            } else {
                ((float*)Cout)[(size_t)(rbase + r) * 128 + col] = v;
            }
        }
    }
}

// ---------------------------------------------------------------------------
// Per-layer state init: m_enc = -inf encoding, s = 0, rst = 0
// ---------------------------------------------------------------------------
__global__ void init_layer_k(int* __restrict__ menc, float* __restrict__ s,
                             float* __restrict__ rst, int n_nodes) {
    int i = blockIdx.x * blockDim.x + threadIdx.x;
    if (i < n_nodes * 4) { menc[i] = (int)0x80000000; s[i] = 0.0f; }
    if (i < n_nodes * 128) rst[i] = 0.0f;
}

// ---------------------------------------------------------------------------
// el[n,h] = feat[n,h,:] . al[h]   er likewise
// ---------------------------------------------------------------------------
__global__ void node_scores_k(const float* __restrict__ feat, const float* __restrict__ al,
                              const float* __restrict__ ar, float* __restrict__ el,
                              float* __restrict__ er, int n_nodes) {
    int i = blockIdx.x * blockDim.x + threadIdx.x;
    if (i >= n_nodes * 4) return;
    int n = i >> 2, h = i & 3;
    const float* f = feat + (size_t)n * 128 + h * 32;
    const float* a = al + h * 32;
    const float* b = ar + h * 32;
    float sl = 0.0f, sr = 0.0f;
#pragma unroll
    for (int d = 0; d < 32; ++d) { float fv = f[d]; sl += fv * a[d]; sr += fv * b[d]; }
    el[i] = sl; er[i] = sr;
}

// monotone float<->int encoding for atomic max on floats
__device__ __forceinline__ int   enc_f(float f) { int i = __float_as_int(f); return i >= 0 ? i : (i ^ 0x7fffffff); }
__device__ __forceinline__ float dec_f(int i)   { return __int_as_float(i >= 0 ? i : (i ^ 0x7fffffff)); }

// ---------------------------------------------------------------------------
// e = leaky_relu(el[src]+er[dst]); segment max via atomic max (int-encoded)
// ---------------------------------------------------------------------------
__global__ void edge_max_k(const int* __restrict__ src, const int* __restrict__ dst,
                           const float* __restrict__ el, const float* __restrict__ er,
                           float* __restrict__ ebuf, int* __restrict__ menc, int E_) {
    int i = blockIdx.x * blockDim.x + threadIdx.x;
    if (i >= E_ * 4) return;
    int e = i >> 2, h = i & 3;
    int sn = src[e], dn = dst[e];
    float v = el[sn * 4 + h] + er[dn * 4 + h];
    v = v > 0.0f ? v : 0.2f * v;
    ebuf[i] = v;
    atomicMax(&menc[dn * 4 + h], enc_f(v));
}

// ---------------------------------------------------------------------------
// a = exp(e - m[dst]); segment sum via native f32 atomic add
// ---------------------------------------------------------------------------
__global__ void edge_exp_k(const int* __restrict__ dst, float* __restrict__ ebuf,
                           const int* __restrict__ menc, float* __restrict__ s, int E_) {
    int i = blockIdx.x * blockDim.x + threadIdx.x;
    if (i >= E_ * 4) return;
    int e = i >> 2, h = i & 3;
    int d4 = dst[e] * 4 + h;
    float a = __expf(ebuf[i] - dec_f(menc[d4]));
    ebuf[i] = a;
    unsafeAtomicAdd(&s[d4], a);
}

__global__ void recip_k(float* __restrict__ s, int n) {
    int i = blockIdx.x * blockDim.x + threadIdx.x;
    if (i < n) s[i] = 1.0f / s[i];
}

// ---------------------------------------------------------------------------
// rst[dst,c] += feat[src,c] * alpha  — 128 lanes per edge: coalesced row
// gather + contiguous atomics (land in L2, feat fits in 192MB L2)
// ---------------------------------------------------------------------------
__global__ void edge_scatter_k(const int* __restrict__ src, const int* __restrict__ dst,
                               const float* __restrict__ feat, const float* __restrict__ ebuf,
                               const float* __restrict__ sinv, float* __restrict__ rst, int E_) {
    int i = blockIdx.x * blockDim.x + threadIdx.x;
    if (i >= E_ * 128) return;
    int e = i >> 7, c = i & 127, h = c >> 5;
    int sn = src[e], dn = dst[e];
    float alpha = ebuf[e * 4 + h] * sinv[dn * 4 + h];
    unsafeAtomicAdd(&rst[(size_t)dn * 128 + c], feat[(size_t)sn * 128 + c] * alpha);
}

// ---------------------------------------------------------------------------
// h = elu(rst + bias)  -> f16 for next WMMA GEMM
// ---------------------------------------------------------------------------
__global__ void node_elu_k(const float* __restrict__ rst, const float* __restrict__ bias,
                           _Float16* __restrict__ hq, int n_nodes) {
    int i = blockIdx.x * blockDim.x + threadIdx.x;
    if (i >= n_nodes * 128) return;
    int c = i & 127;
    float v = rst[i] + bias[c];
    v = v > 0.0f ? v : (__expf(v) - 1.0f);
    hq[i] = (_Float16)v;
}

// ---------------------------------------------------------------------------
// out[n,0:2] = h[n,:] @ W_head + b_head  (NOUT=2 -> scalar GEMV)
// ---------------------------------------------------------------------------
__global__ void head_gemv_k(const _Float16* __restrict__ h, const float* __restrict__ W,
                            const float* __restrict__ b, float* __restrict__ out, int n_nodes) {
    int n = blockIdx.x * blockDim.x + threadIdx.x;
    if (n >= n_nodes) return;
    float o0 = b[0], o1 = b[1];
    const _Float16* hr = h + (size_t)n * 128;
#pragma unroll 8
    for (int k = 0; k < 128; ++k) {
        float hv = (float)hr[k];
        o0 += hv * W[k * 2 + 0];
        o1 += hv * W[k * 2 + 1];
    }
    out[n * 2 + 0] = o0;
    out[n * 2 + 1] = o1;
}

// ---------------------------------------------------------------------------
// Launch
// ---------------------------------------------------------------------------
extern "C" void kernel_launch(void* const* d_in, const int* in_sizes, int n_in,
                              void* d_out, int out_size, void* d_ws, size_t ws_size,
                              hipStream_t stream) {
    const float* x      = (const float*)d_in[0];
    const int*   src    = (const int*)d_in[1];
    const int*   dst    = (const int*)d_in[2];
    const float* W_init = (const float*)d_in[3];
    const float* b_init = (const float*)d_in[4];
    const float* Wfc[2]  = { (const float*)d_in[5],  (const float*)d_in[9]  };
    const float* al[2]   = { (const float*)d_in[6],  (const float*)d_in[10] };
    const float* ar[2]   = { (const float*)d_in[7],  (const float*)d_in[11] };
    const float* bias[2] = { (const float*)d_in[8],  (const float*)d_in[12] };
    const float* W_head = (const float*)d_in[13];
    const float* b_head = (const float*)d_in[14];

    const int N = in_sizes[0] / 256;
    const int E = in_sizes[1];

    // workspace carve-out (256B aligned)
    char* p = (char*)d_ws;
    auto alloc = [&](size_t bytes) -> void* {
        void* r = (void*)p;
        p += (bytes + 255) & ~(size_t)255;
        return r;
    };
    _Float16* x16     = (_Float16*)alloc((size_t)N * 256 * 2);
    _Float16* wi16    = (_Float16*)alloc((size_t)256 * 128 * 2);
    _Float16* wf16[2] = { (_Float16*)alloc((size_t)128 * 128 * 2),
                          (_Float16*)alloc((size_t)128 * 128 * 2) };
    _Float16* h16  = (_Float16*)alloc((size_t)N * 128 * 2);
    float*    feat = (float*)alloc((size_t)N * 128 * 4);
    float*    el   = (float*)alloc((size_t)N * 4 * 4);
    float*    er   = (float*)alloc((size_t)N * 4 * 4);
    int*      menc = (int*)alloc((size_t)N * 4 * 4);
    float*    s    = (float*)alloc((size_t)N * 4 * 4);
    float*    ebuf = (float*)alloc((size_t)E * 4 * 4);
    float*    rst  = (float*)alloc((size_t)N * 128 * 4);

    dim3 b256(256);

    // precision down-convert for WMMA
    f32_to_f16_k<<<(N * 256 + 255) / 256, b256, 0, stream>>>(x, x16, N * 256);
    f32_to_f16_k<<<(256 * 128 + 255) / 256, b256, 0, stream>>>(W_init, wi16, 256 * 128);
    f32_to_f16_k<<<(128 * 128 + 255) / 256, b256, 0, stream>>>(Wfc[0], wf16[0], 128 * 128);
    f32_to_f16_k<<<(128 * 128 + 255) / 256, b256, 0, stream>>>(Wfc[1], wf16[1], 128 * 128);

    const int tiles = N / 16;
    const int gblocks = (tiles + 7) / 8;

    // h0 = relu(x @ W_init + b_init)   [WMMA, f16 out]
    gemm_wmma_k<0><<<gblocks, b256, 0, stream>>>(x16, wi16, b_init, (void*)h16, N, 256);

    for (int l = 0; l < 2; ++l) {
        // feat = h @ Wfc  [WMMA, f32 out]
        gemm_wmma_k<1><<<gblocks, b256, 0, stream>>>(h16, wf16[l], nullptr, (void*)feat, N, 128);
        init_layer_k<<<(N * 128 + 255) / 256, b256, 0, stream>>>(menc, s, rst, N);
        node_scores_k<<<(N * 4 + 255) / 256, b256, 0, stream>>>(feat, al[l], ar[l], el, er, N);
        edge_max_k<<<(E * 4 + 255) / 256, b256, 0, stream>>>(src, dst, el, er, ebuf, menc, E);
        edge_exp_k<<<(E * 4 + 255) / 256, b256, 0, stream>>>(dst, ebuf, menc, s, E);
        recip_k<<<(N * 4 + 255) / 256, b256, 0, stream>>>(s, N * 4);
        long long tot = (long long)E * 128;
        edge_scatter_k<<<(unsigned)((tot + 255) / 256), b256, 0, stream>>>(src, dst, feat, ebuf, s, rst, E);
        node_elu_k<<<(N * 128 + 255) / 256, b256, 0, stream>>>(rst, bias[l], h16, N);
    }

    head_gemv_k<<<(N + 255) / 256, b256, 0, stream>>>(h16, W_head, b_head, (float*)d_out, N);
}
